// HybridGNN_32358283608542
// MI455X (gfx1250) — matrix operand order repaired and checked by reading, not compiled, not run on hardware
//
#include <hip/hip_runtime.h>

#define NN 50000
#define EE 800000
#define EA 850000
#define GG 64
#define FLIPNEGINF 0x007FFFFFu

typedef __attribute__((ext_vector_type(16))) __bf16 v16bf;
typedef __attribute__((ext_vector_type(8)))  float  v8f;

// ---------- helpers ----------
__device__ __forceinline__ unsigned int fflip(float f) {
    unsigned int u = __float_as_uint(f);
    return (u & 0x80000000u) ? ~u : (u | 0x80000000u);
}
__device__ __forceinline__ float funflip(unsigned int k) {
    unsigned int u = (k & 0x80000000u) ? (k & 0x7FFFFFFFu) : ~k;
    return __uint_as_float(u);
}
__device__ __forceinline__ unsigned short f2bf(float f) {
    unsigned int u = __float_as_uint(f);
    unsigned int r = u + 0x7FFFu + ((u >> 16) & 1u);   // round-to-nearest-even
    return (unsigned short)(r >> 16);
}

// ---------- utility kernels ----------
__global__ void gnn_fill_u32(unsigned int* __restrict__ p, unsigned int val, long long n) {
    long long i = (long long)blockIdx.x * blockDim.x + threadIdx.x;
    if (i < n) p[i] = val;
}

__global__ void gnn_cvt_bf16(const float* __restrict__ src, unsigned short* __restrict__ dst, long long n) {
    long long i = (long long)blockIdx.x * blockDim.x + threadIdx.x;
    if (i < n) dst[i] = f2bf(src[i]);
}

// ---------- WMMA bf16 GEMM:  C[M,Nd] = A[M,K] * B[K,Nd] + bias ----------
// grid = (ceil(Mtiles/4), Nd/64), block = 128 (4 waves).
// Each wave: one 16-wide n-tile, 4 consecutive 16-row m-tiles; the B fragment
// is loaded once per K-step and reused for 4 WMMAs (weights stay in registers).
template <int K>
__global__ __launch_bounds__(128) void gnn_gemm_bf16(
        const unsigned short* __restrict__ A,   // M x K (bf16 bits)
        const unsigned short* __restrict__ B,   // K x Nd (bf16 bits)
        const float* __restrict__ bias,
        float* __restrict__ C, int Nd, int Mtiles)
{
    const int wave   = threadIdx.x >> 5;
    const int lane   = threadIdx.x & 31;
    const int mtile0 = blockIdx.x * 4;
    const int ntile  = blockIdx.y * 4 + wave;
    const int r16    = lane & 15;
    const int half   = lane >> 4;
    const int ncol   = ntile * 16 + r16;        // B/C column for this lane

    const int nm = (Mtiles - mtile0 < 4) ? (Mtiles - mtile0) : 4;   // wave-uniform

    union U { v16bf v; unsigned int u[8]; };
    v8f cacc[4] = {v8f{}, v8f{}, v8f{}, v8f{}};

#pragma unroll
    for (int kk = 0; kk < K; kk += 32) {
        U b;
#pragma unroll
        for (int v = 0; v < 8; ++v) {
            // B k order used here (contiguous): lanes 0-15 hold K=0..15, lanes 16-31 hold K=16..31
            int kb = kk + 16 * half + 2 * v;
            unsigned int b0 = B[(size_t)kb * Nd + ncol];
            unsigned int b1 = B[(size_t)(kb + 1) * Nd + ncol];
            b.u[v] = b0 | (b1 << 16);
        }
#pragma unroll
        for (int mt = 0; mt < 4; ++mt) {
            if (mt < nm) {                       // uniform branch: EXEC stays all-ones
                const unsigned short* arow = A + (size_t)((mtile0 + mt) * 16 + r16) * K;
                U a;
#pragma unroll
                for (int v = 0; v < 8; ++v) {
                    // A k order used here (contiguous): lanes 0-15 hold K=0..15, lanes 16-31 hold K=16..31 (matches B)
                    int ka = kk + 16 * half + 2 * v;
                    a.u[v] = *(const unsigned int*)(arow + ka);
                }
                cacc[mt] = __builtin_amdgcn_wmma_f32_16x16x32_bf16(
                               false, a.v, false, b.v, (short)0, cacc[mt], false, false);
            }
        }
    }

    const float bs = bias[ncol];
#pragma unroll
    for (int mt = 0; mt < 4; ++mt) {
        if (mt < nm) {
#pragma unroll
            for (int r = 0; r < 8; ++r) {
                int mr = (mtile0 + mt) * 16 + r + 8 * half;   // C: VGPR r -> M=r(+8 upper half)
                C[(size_t)mr * Nd + ncol] = cacc[mt][r] + bs;
            }
        }
    }
}

// ---------- edge pass A: logits + per-dst max ----------
__global__ void gnn_edge_logits(const float* __restrict__ xl, const float* __restrict__ xr,
                                const float* __restrict__ att, const int* __restrict__ ei,
                                float* __restrict__ s, unsigned int* __restrict__ mxb,
                                int Hh, int Cc)
{
    int idx = blockIdx.x * blockDim.x + threadIdx.x;
    if (idx >= EA * Hh) return;
    int e = idx / Hh, h = idx - e * Hh;
    int sn, dn;
    if (e < EE) { sn = ei[e]; dn = ei[EE + e]; } else { sn = dn = e - EE; }
    const int D = Hh * Cc;
    const float4* pl = (const float4*)(xl + (size_t)sn * D + h * Cc);
    const float4* pr = (const float4*)(xr + (size_t)dn * D + h * Cc);
    const float4* pa = (const float4*)(att + h * Cc);
    __builtin_prefetch(pl, 0, 1);
    __builtin_prefetch(pr, 0, 1);
    float acc = 0.f;
    const int C4 = Cc >> 2;
    for (int c = 0; c < C4; ++c) {
        float4 l = pl[c], r = pr[c], w = pa[c];
        float t;
        t = l.x + r.x; acc += w.x * ((t > 0.f) ? t : 0.2f * t);
        t = l.y + r.y; acc += w.y * ((t > 0.f) ? t : 0.2f * t);
        t = l.z + r.z; acc += w.z * ((t > 0.f) ? t : 0.2f * t);
        t = l.w + r.w; acc += w.w * ((t > 0.f) ? t : 0.2f * t);
    }
    s[idx] = acc;
    atomicMax(&mxb[dn * Hh + h], fflip(acc));
}

// ---------- edge pass B: softmax denominator ----------
__global__ void gnn_edge_den(const float* __restrict__ s, const unsigned int* __restrict__ mxb,
                             float* __restrict__ den, const int* __restrict__ ei, int Hh)
{
    int idx = blockIdx.x * blockDim.x + threadIdx.x;
    if (idx >= EA * Hh) return;
    int e = idx / Hh, h = idx - e * Hh;
    int dn = (e < EE) ? ei[EE + e] : (e - EE);
    float mx = funflip(mxb[dn * Hh + h]);
    atomicAdd(&den[dn * Hh + h], __expf(s[idx] - mx));
}

// ---------- edge pass C: alpha-weighted scatter-add ----------
__global__ void gnn_edge_accum(const float* __restrict__ xl, const float* __restrict__ s,
                               const unsigned int* __restrict__ mxb, const float* __restrict__ den,
                               float* __restrict__ acc, const int* __restrict__ ei,
                               int Hh, int Cc)
{
    int idx = blockIdx.x * blockDim.x + threadIdx.x;
    if (idx >= EA * Hh) return;
    int e = idx / Hh, h = idx - e * Hh;
    int sn, dn;
    if (e < EE) { sn = ei[e]; dn = ei[EE + e]; } else { sn = dn = e - EE; }
    const int D = Hh * Cc;
    float alpha = __expf(s[idx] - funflip(mxb[dn * Hh + h])) / den[dn * Hh + h];
    const float4* pl = (const float4*)(xl + (size_t)sn * D + h * Cc);
    float* pd = acc + (size_t)dn * D + h * Cc;
    __builtin_prefetch(pl, 0, 1);
    const int C4 = Cc >> 2;
    for (int c = 0; c < C4; ++c) {
        float4 l = pl[c];
        atomicAdd(&pd[4 * c + 0], alpha * l.x);
        atomicAdd(&pd[4 * c + 1], alpha * l.y);
        atomicAdd(&pd[4 * c + 2], alpha * l.z);
        atomicAdd(&pd[4 * c + 3], alpha * l.w);
    }
}

// ---------- bias + BN (+ELU) finalize ----------
__global__ void gnn_finalize(const float* __restrict__ acc, const float* __restrict__ cbias,
                             const float* __restrict__ g, const float* __restrict__ be,
                             const float* __restrict__ mrun, const float* __restrict__ vrun,
                             unsigned short* __restrict__ out_bf, float* __restrict__ out_f32,
                             int D, int do_elu)
{
    long long idx = (long long)blockIdx.x * blockDim.x + threadIdx.x;
    if (idx >= (long long)NN * D) return;
    int d = (int)(idx % D);
    float x = acc[idx] + cbias[d];
    x = (x - mrun[d]) * g[d] * rsqrtf(vrun[d] + 1e-5f) + be[d];
    if (do_elu) x = (x > 0.f) ? x : (__expf(x) - 1.f);
    if (out_bf)  out_bf[idx]  = f2bf(x);
    if (out_f32) out_f32[idx] = x;
}

// ---------- graph pooling (mean-acc + max) ----------
__global__ void gnn_pool(const float* __restrict__ h, const int* __restrict__ batch,
                         float* __restrict__ macc, unsigned int* __restrict__ gmx,
                         float* __restrict__ cnt)
{
    long long idx = (long long)blockIdx.x * blockDim.x + threadIdx.x;
    if (idx >= (long long)NN * 128) return;
    int n = (int)(idx >> 7), d = (int)(idx & 127);
    int b = batch[n];
    float v = h[idx];
    atomicAdd(&macc[b * 128 + d], v);
    atomicMax(&gmx[b * 128 + d], fflip(v));
    if (d == 0) atomicAdd(&cnt[b], 1.f);
}

// ---------- classifier MLP: one block per graph ----------
__global__ __launch_bounds__(256) void gnn_mlp(
        const float* __restrict__ macc, const unsigned int* __restrict__ gmx,
        const float* __restrict__ cnt,
        const float* __restrict__ cw0, const float* __restrict__ cb0,
        const float* __restrict__ cw1, const float* __restrict__ cb1,
        const float* __restrict__ cw2, const float* __restrict__ cb2,
        float* __restrict__ out)
{
    __shared__ float z[256], z1[128], z2[64];
    int g = blockIdx.x, t = threadIdx.x;
    if (t < 128) {
        float c = cnt[g]; c = (c > 1.f) ? c : 1.f;
        z[t] = macc[g * 128 + t] / c;
    } else {
        int d = t - 128;
        float mx = funflip(gmx[g * 128 + d]);
        unsigned int u = __float_as_uint(mx);
        z[128 + d] = (((u >> 23) & 0xFFu) != 0xFFu) ? mx : 0.f;   // isfinite
    }
    __syncthreads();
    if (t < 128) {
        float s = cb0[t];
        for (int i = 0; i < 256; ++i) s += z[i] * cw0[i * 128 + t];
        z1[t] = (s > 0.f) ? s : 0.f;
    }
    __syncthreads();
    if (t < 64) {
        float s = cb1[t];
        for (int i = 0; i < 128; ++i) s += z1[i] * cw1[i * 64 + t];
        z2[t] = (s > 0.f) ? s : 0.f;
    }
    __syncthreads();
    if (t == 0) {
        float s = cb2[0];
        for (int i = 0; i < 64; ++i) s += z2[i] * cw2[i];
        out[g] = s;
    }
}

// ---------- host ----------
extern "C" void kernel_launch(void* const* d_in, const int* in_sizes, int n_in,
                              void* d_out, int out_size, void* d_ws, size_t ws_size,
                              hipStream_t stream)
{
    (void)in_sizes; (void)n_in; (void)out_size; (void)ws_size;
    const float* x     = (const float*)d_in[0];
    const int*   ei    = (const int*)d_in[1];
    const int*   batch = (const int*)d_in[2];
    const float* Wl[3]  = {(const float*)d_in[3],  (const float*)d_in[9],  (const float*)d_in[15]};
    const float* bl[3]  = {(const float*)d_in[4],  (const float*)d_in[10], (const float*)d_in[16]};
    const float* Wr[3]  = {(const float*)d_in[5],  (const float*)d_in[11], (const float*)d_in[17]};
    const float* br[3]  = {(const float*)d_in[6],  (const float*)d_in[12], (const float*)d_in[18]};
    const float* att[3] = {(const float*)d_in[7],  (const float*)d_in[13], (const float*)d_in[19]};
    const float* cbi[3] = {(const float*)d_in[8],  (const float*)d_in[14], (const float*)d_in[20]};
    const float* bg[3]  = {(const float*)d_in[21], (const float*)d_in[25], (const float*)d_in[29]};
    const float* bb[3]  = {(const float*)d_in[22], (const float*)d_in[26], (const float*)d_in[30]};
    const float* bm[3]  = {(const float*)d_in[23], (const float*)d_in[27], (const float*)d_in[31]};
    const float* bv[3]  = {(const float*)d_in[24], (const float*)d_in[28], (const float*)d_in[32]};
    const float* cw0 = (const float*)d_in[33]; const float* cb0 = (const float*)d_in[34];
    const float* cw1 = (const float*)d_in[35]; const float* cb1 = (const float*)d_in[36];
    const float* cw2 = (const float*)d_in[37]; const float* cb2 = (const float*)d_in[38];
    float* out = (float*)d_out;

    char* base = (char*)d_ws; size_t off = 0;
    auto alloc = [&](size_t bytes) -> void* {
        void* p = base + off;
        off += (bytes + 255) & ~(size_t)255;
        return p;
    };
    unsigned short* xbf = (unsigned short*)alloc((size_t)NN * 256 * 2);
    const int wsz[3] = {128 * 256, 256 * 256, 256 * 128};
    unsigned short *wlbf[3], *wrbf[3];
    for (int l = 0; l < 3; ++l) {
        wlbf[l] = (unsigned short*)alloc((size_t)wsz[l] * 2);
        wrbf[l] = (unsigned short*)alloc((size_t)wsz[l] * 2);
    }
    float*        xl   = (float*)alloc((size_t)NN * 256 * 4);
    float*        xr   = (float*)alloc((size_t)NN * 256 * 4);
    float*        sb   = (float*)alloc((size_t)EA * 4 * 4);
    unsigned int* mxb  = (unsigned int*)alloc((size_t)NN * 4 * 4);
    float*        den  = (float*)alloc((size_t)NN * 4 * 4);
    float*        acc  = (float*)alloc((size_t)NN * 256 * 4);
    float*        macc = (float*)alloc((size_t)GG * 128 * 4);
    unsigned int* gmx  = (unsigned int*)alloc((size_t)GG * 128 * 4);
    float*        cnt  = (float*)alloc((size_t)GG * 4);
    float*        hout = xr;   // xr is dead after the layer-2 edge_logits pass

    dim3 b256(256);
    auto gs = [](long long n) { return dim3((unsigned)((n + 255) / 256)); };

    // weights + input to bf16
    for (int l = 0; l < 3; ++l) {
        gnn_cvt_bf16<<<gs(wsz[l]), b256, 0, stream>>>(Wl[l], wlbf[l], wsz[l]);
        gnn_cvt_bf16<<<gs(wsz[l]), b256, 0, stream>>>(Wr[l], wrbf[l], wsz[l]);
    }
    gnn_cvt_bf16<<<gs((long long)NN * 128), b256, 0, stream>>>(x, xbf, (long long)NN * 128);

    const int Mtiles = NN / 16;                     // 3125
    for (int l = 0; l < 3; ++l) {
        const int Din  = (l == 0) ? 128 : 256;
        const int Dout = (l == 2) ? 128 : 256;
        const int Hh   = (l == 2) ? 1 : 4;
        const int Cc   = (l == 2) ? 128 : 64;

        dim3 ggrid((Mtiles + 3) / 4, Dout / 64), gblock(128);
        if (Din == 128) {
            gnn_gemm_bf16<128><<<ggrid, gblock, 0, stream>>>(xbf, wlbf[l], bl[l], xl, Dout, Mtiles);
            gnn_gemm_bf16<128><<<ggrid, gblock, 0, stream>>>(xbf, wrbf[l], br[l], xr, Dout, Mtiles);
        } else {
            gnn_gemm_bf16<256><<<ggrid, gblock, 0, stream>>>(xbf, wlbf[l], bl[l], xl, Dout, Mtiles);
            gnn_gemm_bf16<256><<<ggrid, gblock, 0, stream>>>(xbf, wrbf[l], br[l], xr, Dout, Mtiles);
        }

        long long nmx  = (long long)NN * Hh;
        long long nacc = (long long)NN * Dout;
        long long nEH  = (long long)EA * Hh;
        gnn_fill_u32<<<gs(nmx),  b256, 0, stream>>>(mxb, FLIPNEGINF, nmx);
        gnn_fill_u32<<<gs(nmx),  b256, 0, stream>>>((unsigned int*)den, 0u, nmx);
        gnn_fill_u32<<<gs(nacc), b256, 0, stream>>>((unsigned int*)acc, 0u, nacc);

        gnn_edge_logits<<<gs(nEH), b256, 0, stream>>>(xl, xr, att[l], ei, sb, mxb, Hh, Cc);
        gnn_edge_den   <<<gs(nEH), b256, 0, stream>>>(sb, mxb, den, ei, Hh);
        gnn_edge_accum <<<gs(nEH), b256, 0, stream>>>(xl, sb, mxb, den, acc, ei, Hh, Cc);

        if (l < 2)
            gnn_finalize<<<gs(nacc), b256, 0, stream>>>(acc, cbi[l], bg[l], bb[l], bm[l], bv[l],
                                                        xbf, (float*)nullptr, Dout, 1);
        else
            gnn_finalize<<<gs(nacc), b256, 0, stream>>>(acc, cbi[l], bg[l], bb[l], bm[l], bv[l],
                                                        (unsigned short*)nullptr, hout, Dout, 0);
    }

    gnn_fill_u32<<<gs(GG * 128), b256, 0, stream>>>((unsigned int*)macc, 0u, GG * 128);
    gnn_fill_u32<<<gs(GG * 128), b256, 0, stream>>>(gmx, FLIPNEGINF, GG * 128);
    gnn_fill_u32<<<gs(GG),       b256, 0, stream>>>((unsigned int*)cnt, 0u, GG);
    gnn_pool<<<gs((long long)NN * 128), b256, 0, stream>>>(hout, batch, macc, gmx, cnt);
    gnn_mlp<<<dim3(GG), dim3(256), 0, stream>>>(macc, gmx, cnt, cw0, cb0, cw1, cb1, cw2, cb2, out);
}
